// _GNNModel_22703197127250
// MI455X (gfx1250) — compile-verified, matrix operand order checked by reading, not run on hardware
//
#include <hip/hip_runtime.h>

#define N_NODES  100000
#define N_EDGES  1600000
#define N_GRAPHS 1024
#define D_HID    64
#define KDIM     70      // fan_in = 64 + 6
#define KPAD     72      // agg row stride in floats
#define KPAD16   96      // K padded for 3 x (16x16x32) WMMA steps
#define EPSV     1e-5f
#define SENT     -1e30f

typedef __attribute__((ext_vector_type(16))) _Float16 v16h;
typedef __attribute__((ext_vector_type(8)))  _Float16 v8h;
typedef __attribute__((ext_vector_type(8)))  float    v8f;

// Float atomic max via integer atomics (native global_atomic_max_i32 / min_u32,
// avoids CAS loops). Valid for any interleaving given a very-negative init.
__device__ __forceinline__ void atomicMaxF(float* addr, float v) {
  if (!(v < 0.0f)) {
    atomicMax((int*)addr, __float_as_int(v));
  } else {
    atomicMin((unsigned int*)addr, __float_as_uint(v));
  }
}

// ---------------------------------------------------------------- init
__global__ void init_kernel(float* __restrict__ agg, long aggN,
                            float* __restrict__ stats,
                            float* __restrict__ pooled) {
  long i = (long)blockIdx.x * 256 + threadIdx.x;
  if (i < aggN) agg[i] = SENT;
  if (i < 2 * D_HID) stats[i] = 0.0f;
  if (pooled != nullptr && i < (long)N_GRAPHS * D_HID) pooled[i] = SENT;
}

// ------------------------------------------------------- edge scatter-max
// msg = concat(x[src], edge_attr); agg[dst] = segment_max(msg)
__global__ void scatter_kernel(const float* __restrict__ x,
                               const int*   __restrict__ src,
                               const int*   __restrict__ dst,
                               const float* __restrict__ ea,
                               float* __restrict__ agg) {
  unsigned gid = blockIdx.x * 256u + threadIdx.x;
  if (gid >= (unsigned)N_EDGES * KDIM) return;
  unsigned e = gid / KDIM;
  unsigned f = gid - e * KDIM;
  float v;
  if (f < D_HID) {
    v = x[(long)src[e] * D_HID + f];
  } else {
    v = ea[(long)e * 6 + (f - D_HID)];
  }
  atomicMaxF(&agg[(long)dst[e] * KPAD + f], v);
}

// ------------------------------------------------- WMMA GEMM + bias + stats
// h[N,64] = fix(agg[N,70]) @ W[70,64] + b ; stats = per-feature {sum, sumsq}
__global__ void __launch_bounds__(128)
gemm_kernel(const float* __restrict__ agg,
            const float* __restrict__ W,
            const float* __restrict__ bias,
            float* __restrict__ h,
            float* __restrict__ stats) {
  __shared__ _Float16 Alds[16 * KPAD16];        // 3 KB  : A tile, row-major K
  __shared__ _Float16 Wt[D_HID * KPAD16];       // 12 KB : W transposed [n][k]

  const int tid = threadIdx.x;
  const int rowBase = blockIdx.x * 16;

  // Stage W transposed (K-contiguous per output column), zero-pad K 70->96.
  for (int idx = tid; idx < D_HID * KPAD16; idx += 128) {
    int n = idx / KPAD16;
    int k = idx - n * KPAD16;
    float wv = (k < KDIM) ? W[k * D_HID + n] : 0.0f;
    Wt[idx] = (_Float16)wv;
  }
  // Stage A tile; map "no contribution" sentinel -> 0 (reference fill).
  for (int idx = tid; idx < 16 * KPAD16; idx += 128) {
    int i = idx / KPAD16;
    int k = idx - i * KPAD16;
    float av = 0.0f;
    if (k < KDIM) {
      float t = agg[(long)(rowBase + i) * KPAD + k];
      av = (t <= -1e29f) ? 0.0f : t;
    }
    Alds[idx] = (_Float16)av;
  }
  __syncthreads();

  const int wave = tid >> 5;        // 4 waves -> 4 x 16-column tiles
  const int lane = tid & 31;
  const int half = lane >> 4;       // 0: K/M-low half, 1: high half
  const int l15  = lane & 15;
  const int colBase = wave * 16;

  v8f c = {};
#pragma unroll
  for (int kk = 0; kk < KPAD16; kk += 32) {
    // Fragment halves are K-contiguous: two aligned 16B LDS loads each.
    int kbase = kk + half * 8;
    const v8h* ap = (const v8h*)&Alds[l15 * KPAD16 + kbase];
    v8h alo = ap[0];
    v8h ahi = ap[2];   // +16 halves (K + 16)
    const v8h* bp = (const v8h*)&Wt[(colBase + l15) * KPAD16 + kbase];
    v8h blo = bp[0];
    v8h bhi = bp[2];
    v16h a, b;
#pragma unroll
    for (int j = 0; j < 8; ++j) {
      a[j] = alo[j]; a[j + 8] = ahi[j];
      b[j] = blo[j]; b[j + 8] = bhi[j];
    }
    c = __builtin_amdgcn_wmma_f32_16x16x32_f16(
        /*neg_a=*/false, a, /*neg_b=*/false, b,
        /*c_mod=*/(short)0, c, /*reuse_a=*/false, /*reuse_b=*/false);
  }

  // Epilogue: bias, store fp32 h, per-feature sum/sumsq for batch-norm stats.
  const int col = colBase + l15;
  const float bcol = bias[col];
  float s = 0.0f, sq = 0.0f;
#pragma unroll
  for (int vg = 0; vg < 8; ++vg) {
    int m = vg + half * 8;
    float hv = c[vg] + bcol;
    h[(long)(rowBase + m) * D_HID + col] = hv;
    s  += hv;
    sq += hv * hv;
  }
  // Lanes l and l^16 hold the same column -> pairwise reduce, then 16 atomics.
  s  += __shfl_xor(s, 16, 32);
  sq += __shfl_xor(sq, 16, 32);
  if (half == 0) {
    atomicAdd(&stats[col], s);
    atomicAdd(&stats[D_HID + col], sq);
  }
}

// ---------------------------------------- normalize + relu (+ graph pooling)
__global__ void norm_kernel(const float* __restrict__ h,
                            const float* __restrict__ stats,
                            const float* __restrict__ g,
                            const float* __restrict__ beta,
                            float* __restrict__ xout,
                            const int* __restrict__ batch,
                            float* __restrict__ pooled,
                            int doPool) {
  long i = (long)blockIdx.x * 256 + threadIdx.x;
  if (i >= (long)N_NODES * D_HID) return;
  int f = (int)(i & (D_HID - 1));
  float mean = stats[f] * (1.0f / N_NODES);
  float var  = stats[D_HID + f] * (1.0f / N_NODES) - mean * mean;
  float r = rsqrtf(var + EPSV);
  float y = (h[i] - mean) * r * g[f] + beta[f];
  y = fmaxf(y, 0.0f);
  xout[i] = y;
  if (doPool) {
    int node = (int)(i >> 6);
    atomicMaxF(&pooled[(long)batch[node] * D_HID + f], y);
  }
}

// ----------------------------------------------------------- final readout
__global__ void final_kernel(const float* __restrict__ pooled,
                             const float* __restrict__ lw,
                             const float* __restrict__ lb,
                             float* __restrict__ out) {
  int gidx = blockIdx.x * 256 + threadIdx.x;
  if (gidx >= N_GRAPHS) return;
  float acc = 0.0f;
#pragma unroll
  for (int f = 0; f < D_HID; ++f) {
    float p = pooled[gidx * D_HID + f];
    p = (p <= -1e29f) ? 0.0f : p;   // empty-graph fill -> 0
    acc += p * lw[f];
  }
  out[gidx] = acc + lb[0];
}

extern "C" void kernel_launch(void* const* d_in, const int* in_sizes, int n_in,
                              void* d_out, int out_size, void* d_ws, size_t ws_size,
                              hipStream_t stream) {
  (void)in_sizes; (void)n_in; (void)out_size; (void)ws_size;

  const float* x     = (const float*)d_in[0];
  const int*   ei    = (const int*)  d_in[1];   // [2, E]
  const float* ea    = (const float*)d_in[2];
  const int*   batch = (const int*)  d_in[3];
  const float* Wp[3]   = { (const float*)d_in[4],  (const float*)d_in[8],  (const float*)d_in[12] };
  const float* bp[3]   = { (const float*)d_in[5],  (const float*)d_in[9],  (const float*)d_in[13] };
  const float* gp[3]   = { (const float*)d_in[6],  (const float*)d_in[10], (const float*)d_in[14] };
  const float* betap[3]= { (const float*)d_in[7],  (const float*)d_in[11], (const float*)d_in[15] };
  const float* lw = (const float*)d_in[16];
  const float* lb = (const float*)d_in[17];

  // Workspace layout (all fp32):
  float* agg    = (float*)d_ws;                               // N*KPAD   (28.8 MB)
  float* h      = agg    + (size_t)N_NODES * KPAD;            // N*64     (25.6 MB)
  float* xbuf   = h      + (size_t)N_NODES * D_HID;           // N*64     (25.6 MB)
  float* pooled = xbuf   + (size_t)N_NODES * D_HID;           // G*64     (256 KB)
  float* stats  = pooled + (size_t)N_GRAPHS * D_HID;          // 128 floats

  const int* src = ei;
  const int* dst = ei + N_EDGES;

  const long aggN = (long)N_NODES * KPAD;
  const unsigned initBlocks = (unsigned)((aggN + 255) / 256);
  const long scatN = (long)N_EDGES * KDIM;
  const unsigned scatBlocks = (unsigned)((scatN + 255) / 256);
  const unsigned normBlocks = (unsigned)(((long)N_NODES * D_HID + 255) / 256);

  const float* xin = x;
  for (int L = 0; L < 3; ++L) {
    init_kernel<<<initBlocks, 256, 0, stream>>>(agg, aggN, stats,
                                                (L == 0) ? pooled : nullptr);
    scatter_kernel<<<scatBlocks, 256, 0, stream>>>(xin, src, dst, ea, agg);
    gemm_kernel<<<N_NODES / 16, 128, 0, stream>>>(agg, Wp[L], bp[L], h, stats);
    norm_kernel<<<normBlocks, 256, 0, stream>>>(h, stats, gp[L], betap[L],
                                                xbuf, batch, pooled, (L == 2) ? 1 : 0);
    xin = xbuf;
  }
  final_kernel<<<(N_GRAPHS + 255) / 256, 256, 0, stream>>>(pooled, lw, lb, (float*)d_out);
}
